// Stage_1_Module_71055938945222
// MI455X (gfx1250) — matrix operand order repaired and checked by reading, not compiled, not run on hardware
//
#include <hip/hip_runtime.h>
#include <cmath>

typedef __attribute__((ext_vector_type(16))) _Float16 v16h;
typedef __attribute__((ext_vector_type(4)))  _Float16 v4h;
typedef __attribute__((ext_vector_type(8)))  float    v8f;

#define CC   64        // n_feats
#define DGRP 8         // deformable groups
#define BB   2
#define HH   128
#define WW   128
#define HWW  (HH * WW) // 16384

// ---------------------------------------------------------------------------
// Bilinear x2 upsample, align_corners=True (matches reference _up2)
// ---------------------------------------------------------------------------
__global__ void up2_kernel(const float* __restrict__ in, float* __restrict__ out) {
    constexpr int h = HH / 2, w = WW / 2;
    int idx = blockIdx.x * blockDim.x + threadIdx.x;
    if (idx >= BB * CC * HH * WW) return;
    int ox = idx & (WW - 1);
    int oy = (idx >> 7) & (HH - 1);
    int bc = idx >> 14;                       // b*CC + c
    float cy = oy * ((float)(h - 1) / (float)(HH - 1));
    float cx = ox * ((float)(w - 1) / (float)(WW - 1));
    int y0 = (int)floorf(cy); int x0 = (int)floorf(cx);
    int y1 = min(y0 + 1, h - 1); int x1 = min(x0 + 1, w - 1);
    float fy = cy - (float)y0, fx = cx - (float)x0;
    const float* p = in + (size_t)bc * h * w;
    float v00 = p[y0 * w + x0], v01 = p[y0 * w + x1];
    float v10 = p[y1 * w + x0], v11 = p[y1 * w + x1];
    float t0 = v00 * (1.f - fy) + v10 * fy;
    float t1 = v01 * (1.f - fy) + v11 * fy;
    out[idx] = t0 * (1.f - fx) + t1 * fx;
}

// ---------------------------------------------------------------------------
// Implicit-GEMM 3x3 conv (pad=1) via v_wmma_f32_16x16x32_f16.
// Rolled reduction loop (one 32-wide K-chunk in flight: ~80 batched loads)
// to keep VGPR use low -> multi-wave occupancy, no spills.
// ---------------------------------------------------------------------------
template <int CINA, int CINB, int COUT, int ACT>
__global__ void __launch_bounds__(32)
conv3x3_wmma(const float* __restrict__ inA, const float* __restrict__ inB,
             float scaleB, const float* __restrict__ wgt,
             const float* __restrict__ bias, float* __restrict__ out) {
    constexpr int  CIN    = CINA + CINB;
    constexpr int  NCH    = CIN / 32;            // chunks per tap (2 or 4)
    constexpr int  LSH    = (NCH == 4) ? 2 : 1;
    constexpr bool MGUARD = (COUT % 16) != 0;    // Cout not tile-aligned (216)

    const int lane = threadIdx.x;
    const int hi   = lane >> 4;          // 0: lanes 0-15, 1: lanes 16-31
    const int lid  = lane & 15;
    const int n0   = blockIdx.x * 64;    // pixel tile (same row: WW % 64 == 0)
    const int m0   = blockIdx.y * 16;

    const int b   = n0 >> 14;            // / HWW
    const int rem = n0 & (HWW - 1);
    const int y   = rem >> 7;            // / WW
    const int x0  = rem & (WW - 1);

    const int   m   = m0 + lid;
    const int   mcl = MGUARD ? min(m, COUT - 1) : m;
    const float msc = (!MGUARD || m < COUT) ? 1.f : 0.f;

    v8f acc[4] = {};

    #pragma unroll 1
    for (int tc = 0; tc < 9 * NCH; ++tc) {
        const int t  = tc >> LSH;                // tap 0..8
        const int cb = (tc & (NCH - 1)) << 5;    // channel chunk base

        const int  dy  = t / 3 - 1, dx = t % 3 - 1;
        const int  iy  = y + dy;
        const bool yv  = (iy >= 0) && (iy < HH);
        const int  iyc = min(max(iy, 0), HH - 1);

        // ---- A fragment: weights [16 x 32] f16 (contiguous k = hi*16 + e) ----
        v16h afrag;
        const float* wp = wgt + ((size_t)mcl * CIN + cb + hi * 16) * 9 + t;
        #pragma unroll
        for (int e = 0; e < 16; ++e) {
            const int k = e;                               // compile-time
            afrag[e] = (_Float16)(wp[k * 9] * msc);
        }

        // which tensor feeds this chunk (uniform scalar select)
        const bool   fromA = (cb < CINA);
        const int    cbl   = fromA ? cb : (cb - CINA);
        const float  csc   = fromA ? 1.f : scaleB;
        const float* src   = fromA ? inA : inB;
        const int    csrc  = fromA ? CINA : CINB;

        // ---- 4 B fragments (4 x 16 pixels), reuse A ----
        #pragma unroll
        for (int s = 0; s < 4; ++s) {
            const int  xx  = x0 + s * 16 + lid;
            const int  ix  = xx + dx;
            const bool pv  = yv && (ix >= 0) && (ix < WW);
            const int  ixc = min(max(ix, 0), WW - 1);
            const float sm = pv ? csc : 0.f;               // zero-pad via mask
            const float* p = src +
                ((size_t)(b * csrc + cbl + hi * 16) * HH + iyc) * WW + ixc;
            v16h bfrag;
            #pragma unroll
            for (int e = 0; e < 16; ++e)                   // imm offsets e*64KiB
                bfrag[e] = (_Float16)(p[e * HWW] * sm);
            acc[s] = __builtin_amdgcn_wmma_f32_16x16x32_f16(
                false, afrag, false, bfrag, (short)0, acc[s], false, false);
        }
    }

    // ---- epilogue: bias + (optional) leaky-ReLU ----
    #pragma unroll
    for (int s = 0; s < 4; ++s) {
        const int xx = x0 + s * 16 + lid;
        #pragma unroll
        for (int r = 0; r < 8; ++r) {
            const int mr = m0 + r + (hi ? 8 : 0);
            if (!MGUARD || mr < COUT) {
                float v = acc[s][r] + bias[MGUARD ? min(mr, COUT - 1) : mr];
                if (ACT) v = (v >= 0.f) ? v : 0.1f * v;
                out[((size_t)(b * COUT + mr) * HH + y) * WW + xx] = v;
            }
        }
    }
}

// ---------------------------------------------------------------------------
// Modulated deformable conv (DCNv2), stride=1 pad=1 dil=1.
// co (216 ch): [0:72)=off_y, [72:144)=off_x, [144:216)=mask logits (sigmoid
// fused). One wave per 16-pixel tile.
//   phase 1: modulated bilinear im2col -> LDS [16 pixels][576+pad K] f16,
//            4 halves packed per ds_store_b64, conflict-free stride (292%64=36)
//   phase 2: 64x16 GEMM, K=576, rolled 18 K-chunks; B fragment = one 32-byte
//            contiguous LDS vector load; t = kc>>1 constant per chunk.
// ---------------------------------------------------------------------------
#define KROW 584   // 576 + 8 halves pad (1168 B row stride)

__global__ void __launch_bounds__(32)
dcn_wmma(const float* __restrict__ x, const float* __restrict__ co,
         const float* __restrict__ wgt, const float* __restrict__ bias,
         float* __restrict__ out) {
    __shared__ _Float16 val[16 * KROW];

    const int lane = threadIdx.x;
    const int hi   = lane >> 4;
    const int lid  = lane & 15;
    const int n0   = blockIdx.x * 16;
    const int b    = n0 >> 14;
    const int rem  = n0 & (HWW - 1);
    const int y    = rem >> 7;
    const int xx   = (rem & (WW - 1)) + lid;

    const size_t coB    = (size_t)b * 216 * HWW;
    const size_t pixoff = (size_t)y * WW + xx;

    // ---- phase 1: modulated bilinear sampling into LDS ----
    #pragma unroll 1
    for (int g = 0; g < DGRP; ++g) {
        #pragma unroll 1
        for (int t = 0; t < 9; ++t) {
            const int ch = g * 9 + t;
            const float offy = co[coB + (size_t)ch * HWW + pixoff];
            const float offx = co[coB + (size_t)(72 + ch) * HWW + pixoff];
            const float mraw = co[coB + (size_t)(144 + ch) * HWW + pixoff];
            const float mk   = 1.f / (1.f + __expf(-mraw));

            const float py = offy + (float)(t / 3 - 1) + (float)y;
            const float px = offx + (float)(t % 3 - 1) + (float)xx;
            const float y0f = floorf(py), x0f = floorf(px);
            const float wy = py - y0f, wx = px - x0f;
            const int iy0 = (int)y0f, ix0 = (int)x0f;
            const int iy1 = iy0 + 1,  ix1 = ix0 + 1;
            const bool vy0 = (iy0 >= 0) && (iy0 < HH);
            const bool vy1 = (iy1 >= 0) && (iy1 < HH);
            const bool vx0 = (ix0 >= 0) && (ix0 < WW);
            const bool vx1 = (ix1 >= 0) && (ix1 < WW);
            const int cy0 = min(max(iy0, 0), HH - 1);
            const int cy1 = min(max(iy1, 0), HH - 1);
            const int cx0 = min(max(ix0, 0), WW - 1);
            const int cx1 = min(max(ix1, 0), WW - 1);
            const float w00 = (1.f - wy) * (1.f - wx) * ((vy0 && vx0) ? 1.f : 0.f);
            const float w01 = (1.f - wy) * wx         * ((vy0 && vx1) ? 1.f : 0.f);
            const float w10 = wy * (1.f - wx)         * ((vy1 && vx0) ? 1.f : 0.f);
            const float w11 = wy * wx                 * ((vy1 && vx1) ? 1.f : 0.f);
            const int a00 = cy0 * WW + cx0, a01 = cy0 * WW + cx1;
            const int a10 = cy1 * WW + cx0, a11 = cy1 * WW + cx1;

            // half-wave splits the 8 channels of this group; 4 consecutive
            // channels -> one packed ds_store_b64 per lane
            const float* xp = x + (size_t)(b * CC + g * 8 + hi * 4) * HWW;
            v4h pk;
            #pragma unroll
            for (int c = 0; c < 4; ++c) {
                const float sv = w00 * xp[c * HWW + a00] + w01 * xp[c * HWW + a01]
                               + w10 * xp[c * HWW + a10] + w11 * xp[c * HWW + a11];
                pk[c] = (_Float16)(sv * mk);
            }
            *(v4h*)(&val[lid * KROW + t * CC + g * 8 + hi * 4]) = pk;
        }
    }
    __syncthreads();

    // ---- phase 2: GEMM 64(out-ch) x 16(pixels), K = 576, r = t*64 + ci ----
    v8f acc[4] = {};
    #pragma unroll 1
    for (int kc = 0; kc < 18; ++kc) {
        // B fragment: 16 contiguous halves (32 bytes) from this lane's row
        const int kb = kc * 32 + hi * 16;
        const v16h bfrag = *(const v16h*)(&val[lid * KROW + kb]);

        // A fragments: t = kc>>1 constant per chunk; ci = (kc&1)*32 + k
        const int t    = kc >> 1;
        const int cilo = (kc & 1) << 5;
        #pragma unroll
        for (int mt = 0; mt < 4; ++mt) {
            const int m = mt * 16 + lid;
            const float* wp = wgt + (size_t)m * (CC * 9) + (cilo + hi * 16) * 9 + t;
            v16h afrag;
            #pragma unroll
            for (int e = 0; e < 16; ++e) {
                const int k = e;                           // imm offsets k*36B
                afrag[e] = (_Float16)wp[k * 9];
            }
            acc[mt] = __builtin_amdgcn_wmma_f32_16x16x32_f16(
                false, afrag, false, bfrag, (short)0, acc[mt], false, false);
        }
    }

    #pragma unroll
    for (int mt = 0; mt < 4; ++mt) {
        #pragma unroll
        for (int r = 0; r < 8; ++r) {
            const int m = mt * 16 + r + (hi ? 8 : 0);
            out[((size_t)(b * CC + m) * HH + y) * WW + xx] = acc[mt][r] + bias[m];
        }
    }
}

// ---------------------------------------------------------------------------
extern "C" void kernel_launch(void* const* d_in, const int* in_sizes, int n_in,
                              void* d_out, int out_size, void* d_ws, size_t ws_size,
                              hipStream_t stream) {
    (void)in_sizes; (void)n_in; (void)out_size; (void)ws_size;

    const float* x        = (const float*)d_in[0];
    const float* yv       = (const float*)d_in[1];
    const float* prev_off = (const float*)d_in[2];
    const float* prev_fea = (const float*)d_in[3];
    const float* w1 = (const float*)d_in[4];   const float* b1 = (const float*)d_in[5];
    const float* w2 = (const float*)d_in[6];   const float* b2 = (const float*)d_in[7];
    const float* w3 = (const float*)d_in[8];   const float* b3 = (const float*)d_in[9];
    const float* w_off = (const float*)d_in[10]; const float* b_off = (const float*)d_in[11];
    const float* w_dcn = (const float*)d_in[12]; const float* b_dcn = (const float*)d_in[13];
    const float* w_fea = (const float*)d_in[14]; const float* b_fea = (const float*)d_in[15];

    const size_t TEN = (size_t)BB * CC * HH * WW;      // 2,097,152 elems

    float* off_out = (float*)d_out;                    // tuple output 0
    float* out_out = (float*)d_out + TEN;              // tuple output 1

    float* ws = (float*)d_ws;
    float* P0 = ws;                 // up2 buffer (prev_off_up, then prev_fea_up)
    float* P1 = ws + TEN;           // off1, then fea
    float* P2 = ws + 2 * TEN;       // off2 ...
    float* CO = ws + 2 * TEN;       // ... overlaid by co (216 ch) once off2 dead

    const dim3 blk(32);
    const int  ntiles = (BB * HH * WW) / 64;   // 512 pixel tiles of 64
    const int  nelem  = BB * CC * HH * WW;

    // 1) prev_offset = up2(prev_offset)
    up2_kernel<<<(nelem + 255) / 256, 256, 0, stream>>>(prev_off, P0);
    // 2) off = lrelu(conv(cat[x,y], w1))
    conv3x3_wmma<64, 64, 64, 1><<<dim3(ntiles, 4), blk, 0, stream>>>(
        x, yv, 1.f, w1, b1, P1);
    // 3) off = lrelu(conv(cat[off, prev_offset*2], w2))
    conv3x3_wmma<64, 64, 64, 1><<<dim3(ntiles, 4), blk, 0, stream>>>(
        P1, P0, 2.f, w2, b2, P2);
    // 4) off = lrelu(conv(off, w3))  -> tuple output 0
    conv3x3_wmma<64, 0, 64, 1><<<dim3(ntiles, 4), blk, 0, stream>>>(
        P2, nullptr, 1.f, w3, b3, off_out);
    // 5) co = conv(off, w_off)  (216 channels, no activation)
    conv3x3_wmma<64, 0, 216, 0><<<dim3(ntiles, 14), blk, 0, stream>>>(
        off_out, nullptr, 1.f, w_off, b_off, CO);
    // 6) fea = deform_conv(x, offsets, sigmoid(mask), w_dcn)
    dcn_wmma<<<dim3((BB * HH * WW) / 16), blk, 0, stream>>>(
        x, CO, w_dcn, b_dcn, P1);
    // 7) prev_fea = up2(prev_stage_fea)
    up2_kernel<<<(nelem + 255) / 256, 256, 0, stream>>>(prev_fea, P0);
    // 8) out = lrelu(conv(cat[fea, prev_fea], w_fea))  -> tuple output 1
    conv3x3_wmma<64, 64, 64, 1><<<dim3(ntiles, 4), blk, 0, stream>>>(
        P1, P0, 1.f, w_fea, b_fea, out_out);
}
